// TransformerDecoder_40046275068501
// MI455X (gfx1250) — matrix operand order repaired and checked by reading, not compiled, not run on hardware
//
#include <hip/hip_runtime.h>
#include <hip/hip_bf16.h>
#include <math.h>

typedef unsigned short u16;
typedef __attribute__((ext_vector_type(8)))  __bf16 v8bf;
typedef __attribute__((ext_vector_type(16))) __bf16 v16bf;
typedef __attribute__((ext_vector_type(8)))  float  v8f;

namespace cfg {
constexpr int L = 4, B = 4, N = 2048, D = 512, H = 8, DH = 64, MLP = 2048;
constexpr int BN  = B * N;        // 8192 rows in the flattened token dim
constexpr int QKV = 3 * H * DH;   // 1536
}

// ---------------------------------------------------------------- helpers ---

__device__ __forceinline__ u16 f2bf(float f) {           // RNE float->bf16
  unsigned int u = __float_as_uint(f);
  u += 0x7fffu + ((u >> 16) & 1u);
  return (u16)(u >> 16);
}

__device__ __forceinline__ v8f v8f_zero() {
  v8f r;
#pragma unroll
  for (int e = 0; e < 8; ++e) r[e] = 0.f;
  return r;
}

__device__ __forceinline__ v16bf ld_pair(const u16* p0, const u16* p1) {
  v8bf a = *(const v8bf*)p0;
  v8bf b = *(const v8bf*)p1;
  return __builtin_shufflevector(a, b, 0, 1, 2, 3, 4, 5, 6, 7,
                                 8, 9, 10, 11, 12, 13, 14, 15);
}
// A-fragment (16x32 bf16, ISA layout): lane<16 -> row M=lane, K {k0..k0+7, k0+16..k0+23}
// lane>=16 -> row M=lane-16, K {k0+8..k0+15, k0+24..k0+31}.  p = rowbase + k0 + 8*hi
__device__ __forceinline__ v16bf ldA(const u16* p) { return ld_pair(p, p + 16); }
// B-fragment (32x16 bf16): lane<16 -> col N=lane, K {k0..k0+7, k0+16..k0+23};
// lane>=16 -> col N=lane-16, K {k0+8..k0+15, k0+24..k0+31}.  p = colbase + k0 + 8*hi
__device__ __forceinline__ v16bf ldB(const u16* p) { return ld_pair(p, p + 16); }

__device__ __forceinline__ v8f wmma_bf16(v16bf a, v16bf b, v8f c) {
  return __builtin_amdgcn_wmma_f32_16x16x32_bf16(false, a, false, b,
                                                 (short)0, c, false, false);
}

// CDNA5 async global->LDS copy (16B per lane), tracked by ASYNCcnt.
// LDS generic addresses carry the LDS byte offset in addr[31:0] (ISA 10.2).
__device__ __forceinline__ void async_load_b128(const void* gptr, void* lptr) {
  unsigned lds = (unsigned)(size_t)lptr;
  asm volatile("global_load_async_to_lds_b128 %0, %1, off"
               :
               : "v"(lds), "v"(gptr)
               : "memory");
}
__device__ __forceinline__ void wait_async() {
  asm volatile("s_wait_asynccnt 0" ::: "memory");
}

// ------------------------------------------------------------ prep kernels ---

__global__ void copy4_kernel(const float4* __restrict__ src,
                             float4* __restrict__ dst, int n4) {
  int i = blockIdx.x * 256 + threadIdx.x;
  if (i < n4) dst[i] = src[i];
}

// src f32 [R][C]  ->  dst bf16 [C][R]   (coalesced writes)
__global__ void tconv_kernel(const float* __restrict__ src, u16* __restrict__ dst,
                             int R, int C) {
  size_t i = (size_t)blockIdx.x * 256 + threadIdx.x;
  if (i >= (size_t)R * C) return;
  int r = (int)(i % R);
  int c = (int)(i / R);
  dst[i] = f2bf(src[(size_t)r * C + c]);
}

__global__ __launch_bounds__(256) void ln_kernel(const float* __restrict__ x,
                                                 const float* __restrict__ w,
                                                 const float* __restrict__ bb,
                                                 u16* __restrict__ out) {
  using namespace cfg;
  const int row = blockIdx.x, tid = threadIdx.x;
  const float* xr = x + (size_t)row * D;
  float v0 = xr[tid], v1 = xr[tid + 256];
  __shared__ float r1[256], r2[256];
  r1[tid] = v0 + v1;
  r2[tid] = v0 * v0 + v1 * v1;
  __syncthreads();
  for (int off = 128; off > 0; off >>= 1) {
    if (tid < off) { r1[tid] += r1[tid + off]; r2[tid] += r2[tid + off]; }
    __syncthreads();
  }
  const float mu  = r1[0] * (1.f / D);
  const float var = r2[0] * (1.f / D) - mu * mu;
  const float inv = rsqrtf(var + 1e-5f);
  out[(size_t)row * D + tid]       = f2bf((v0 - mu) * inv * w[tid] + bb[tid]);
  out[(size_t)row * D + tid + 256] = f2bf((v1 - mu) * inv * w[tid + 256] + bb[tid + 256]);
}

// ------------------------------------------------------------- GEMM kernel ---
// C[M,Nn] = A_bf16[M,K] @ W^T_bf16[Nn,K]  (+bias / GELU / residual epilogue)
// EPI 0: store bf16   EPI 1: bias + exact GELU -> bf16   EPI 2: bias + xio += v
template <int EPI>
__global__ __launch_bounds__(256) void gemm_bf16_kernel(
    const u16* __restrict__ A, const u16* __restrict__ WT,
    const float* __restrict__ bias, float* __restrict__ xio,
    u16* __restrict__ obf, int M, int Nn, int K) {
  const int tid  = threadIdx.x;
  const int lane = tid & 31;
  const int wv   = tid >> 5;            // 8 waves: 4 (M) x 2 (N)
  const int wm   = wv >> 1, wn = wv & 1;
  const int hi   = lane >> 4, ln = lane & 15;
  const int m0   = blockIdx.y * 128 + wm * 32;
  const int n0   = blockIdx.x * 128 + wn * 64;

  v8f acc[2][4];
#pragma unroll
  for (int i = 0; i < 2; ++i)
#pragma unroll
    for (int j = 0; j < 4; ++j) acc[i][j] = v8f_zero();

  const u16* ar0 = A + (size_t)(m0 + ln) * K;
  const u16* ar1 = A + (size_t)(m0 + 16 + ln) * K;
  const u16* br[4];
#pragma unroll
  for (int j = 0; j < 4; ++j) br[j] = WT + (size_t)(n0 + j * 16 + ln) * K;

  for (int k0 = 0; k0 < K; k0 += 32) {
    if (k0 + 32 < K) {                  // prefetch next K-step (L2-resident)
      __builtin_prefetch(ar0 + k0 + 32 + hi * 8, 0, 1);
      __builtin_prefetch(ar1 + k0 + 32 + hi * 8, 0, 1);
#pragma unroll
      for (int j = 0; j < 4; ++j)
        __builtin_prefetch(br[j] + k0 + 32 + hi * 16, 0, 1);
    }
    v16bf a0 = ldA(ar0 + k0 + hi * 8);
    v16bf a1 = ldA(ar1 + k0 + hi * 8);
    v16bf b[4];
#pragma unroll
    for (int j = 0; j < 4; ++j) b[j] = ldB(br[j] + k0 + hi * 8);
#pragma unroll
    for (int j = 0; j < 4; ++j) {
      acc[0][j] = wmma_bf16(a0, b[j], acc[0][j]);
      acc[1][j] = wmma_bf16(a1, b[j], acc[1][j]);
    }
  }

#pragma unroll
  for (int i = 0; i < 2; ++i)
#pragma unroll
    for (int j = 0; j < 4; ++j) {
      const int gcol = n0 + j * 16 + ln;
      const float bv = (EPI == 0) ? 0.f : bias[gcol];
#pragma unroll
      for (int e = 0; e < 8; ++e) {                 // C layout: M = e + 8*hi
        const int grow   = m0 + i * 16 + e + hi * 8;
        const size_t idx = (size_t)grow * Nn + gcol;
        float v = acc[i][j][e] + bv;
        if (EPI == 0) {
          obf[idx] = f2bf(v);
        } else if (EPI == 1) {
          obf[idx] = f2bf(0.5f * v * (1.f + erff(v * 0.70710678118654752f)));
        } else {
          xio[idx] += v;
        }
      }
    }
}

// -------------------------------------------------- flash attention kernel ---
// grid (N/64, B*H), 128 threads = 4 waves; wave w owns query rows [w*16, w*16+16)
__global__ __launch_bounds__(128) void attn_kernel(const u16* __restrict__ qkv,
                                                   u16* __restrict__ out) {
  using namespace cfg;
  constexpr int PAD = 8;
  const int qt   = blockIdx.x;
  const int b    = blockIdx.y >> 3;
  const int h    = blockIdx.y & 7;
  const int tid  = threadIdx.x;
  const int lane = tid & 31;
  const int wv   = tid >> 5;
  const int hi   = lane >> 4, ln = lane & 15;

  __shared__ u16   Ksh[64][64 + PAD];   // K tile [key][dh]
  __shared__ u16   Vts[64][64 + PAD];   // V tile transposed [dh][key]
  __shared__ u16   Psh[64][64 + PAD];   // probs bf16 [q][key]
  __shared__ float Ssh[64][64 + PAD];   // raw scores
  __shared__ float row_m[64], row_l[64], row_a[64];

  const size_t bbase = (size_t)b * N * QKV;
  const int q0 = qt * 64;

  // Q fragments (held for the whole kernel)
  v16bf aq[2];
#pragma unroll
  for (int kh = 0; kh < 2; ++kh) {
    const u16* p = qkv + bbase + (size_t)(q0 + wv * 16 + ln) * QKV + h * DH +
                   kh * 32 + hi * 8;
    aq[kh] = ldA(p);
  }

  v8f o[4];
#pragma unroll
  for (int t = 0; t < 4; ++t) o[t] = v8f_zero();
  if (tid < 64) { row_m[tid] = -INFINITY; row_l[tid] = 0.f; }

  const int nkt = qt + 1;                 // causal: only key tiles <= query tile
  for (int kt = 0; kt < nkt; ++kt) {
    __syncthreads();                      // protect K/V tiles from prior readers
#pragma unroll
    for (int i = 0; i < 4; ++i) {         // stage 64x64 K and V^T (8-elem chunks)
      const int c  = tid + i * 128;
      const int r  = c >> 3;
      const int cc = (c & 7) * 8;
      const u16* kp = qkv + bbase + (size_t)(kt * 64 + r) * QKV + H * DH + h * DH + cc;
      const u16* vp = qkv + bbase + (size_t)(kt * 64 + r) * QKV + 2 * H * DH + h * DH + cc;
      async_load_b128(kp, &Ksh[r][cc]);   // K tile: async global->LDS, no VGPR trip
      uint4 vt = *(const uint4*)vp;       // V tile: needs transpose, manual scatter
      const u16* vvp = (const u16*)&vt;
#pragma unroll
      for (int e = 0; e < 8; ++e) Vts[cc + e][r] = vvp[e];
    }
    wait_async();
    __syncthreads();

    // S = (Q @ K^T) * scale   (8 WMMAs per wave)
    v8f s[4];
#pragma unroll
    for (int nt = 0; nt < 4; ++nt) s[nt] = v8f_zero();
#pragma unroll
    for (int kh = 0; kh < 2; ++kh)
#pragma unroll
      for (int nt = 0; nt < 4; ++nt) {
        v16bf bk = ldB(&Ksh[nt * 16 + ln][kh * 32 + hi * 8]);
        s[nt] = wmma_bf16(aq[kh], bk, s[nt]);
      }
#pragma unroll
    for (int nt = 0; nt < 4; ++nt)
#pragma unroll
      for (int e = 0; e < 8; ++e)
        Ssh[wv * 16 + e + hi * 8][nt * 16 + ln] = s[nt][e] * 0.125f;
    __syncthreads();

    // online softmax: one thread per query row (mask = strict upper triangle)
    if (tid < 64) {
      const int r = tid, qrow = q0 + r, kb = kt * 64;
      float m_old = row_m[r], tmax = m_old;
      for (int j = 0; j < 64; ++j) {
        float v = (kb + j > qrow) ? -INFINITY : Ssh[r][j];
        tmax = fmaxf(tmax, v);
      }
      const float m_new = tmax;
      const float alpha = __expf(m_old - m_new);
      float psum = 0.f;
      for (int j = 0; j < 64; ++j) {
        float p = (kb + j > qrow) ? 0.f : __expf(Ssh[r][j] - m_new);
        Psh[r][j] = f2bf(p);
        psum += p;
      }
      row_m[r] = m_new;
      row_a[r] = alpha;
      row_l[r] = alpha * row_l[r] + psum;
    }
    __syncthreads();

    // rescale running O, then O += P @ V  (8 WMMAs per wave)
    float al[8];
#pragma unroll
    for (int e = 0; e < 8; ++e) al[e] = row_a[wv * 16 + e + hi * 8];
#pragma unroll
    for (int t = 0; t < 4; ++t)
#pragma unroll
      for (int e = 0; e < 8; ++e) o[t][e] *= al[e];
#pragma unroll
    for (int kh = 0; kh < 2; ++kh) {
      v16bf ap = ldA(&Psh[wv * 16 + ln][kh * 32 + hi * 8]);
#pragma unroll
      for (int dt = 0; dt < 4; ++dt) {
        v16bf bv = ldB(&Vts[dt * 16 + ln][kh * 32 + hi * 8]);
        o[dt] = wmma_bf16(ap, bv, o[dt]);
      }
    }
  }

  // epilogue: normalize and store bf16
  float inv[8];
#pragma unroll
  for (int e = 0; e < 8; ++e) inv[e] = 1.f / row_l[wv * 16 + e + hi * 8];
#pragma unroll
  for (int dt = 0; dt < 4; ++dt)
#pragma unroll
    for (int e = 0; e < 8; ++e) {
      const int grow = b * N + q0 + wv * 16 + e + hi * 8;
      const int gcol = h * DH + dt * 16 + ln;
      out[(size_t)grow * (H * DH) + gcol] = f2bf(o[dt][e] * inv[e]);
    }
}

// ---------------------------------------------------------------- launcher ---

extern "C" void kernel_launch(void* const* d_in, const int* in_sizes, int n_in,
                              void* d_out, int out_size, void* d_ws, size_t ws_size,
                              hipStream_t stream) {
  using namespace cfg;
  (void)in_sizes; (void)n_in; (void)out_size; (void)ws_size;

  const float* x    = (const float*)d_in[0];
  // d_in[1] = mask: strict upper triangle == causal, handled analytically
  const float* ln1w = (const float*)d_in[2];
  const float* ln1b = (const float*)d_in[3];
  const float* wqkv = (const float*)d_in[4];
  const float* wout = (const float*)d_in[5];
  const float* bout = (const float*)d_in[6];
  const float* ln2w = (const float*)d_in[7];
  const float* ln2b = (const float*)d_in[8];
  const float* w1   = (const float*)d_in[9];
  const float* b1   = (const float*)d_in[10];
  const float* w2   = (const float*)d_in[11];
  const float* b2   = (const float*)d_in[12];
  float* xo = (float*)d_out;               // residual stream, updated in place

  // workspace layout (bf16 elements)
  u16* wqkvT = (u16*)d_ws;
  u16* woutT = wqkvT + (size_t)L * QKV * D;
  u16* w1T   = woutT + (size_t)L * D * D;
  u16* w2T   = w1T   + (size_t)L * MLP * D;
  u16* hbuf  = w2T   + (size_t)L * D * MLP;
  u16* qkvb  = hbuf  + (size_t)BN * D;
  u16* attb  = qkvb  + (size_t)BN * QKV;
  u16* mlpb  = attb  + (size_t)BN * D;

  { // x -> residual stream
    const int n4 = BN * D / 4;
    copy4_kernel<<<dim3((n4 + 255) / 256), 256, 0, stream>>>(
        (const float4*)x, (float4*)xo, n4);
  }
  // weights -> transposed bf16 (W^T[n][k]) so B-fragments are contiguous
  for (int l = 0; l < L; ++l) {
    tconv_kernel<<<dim3((D * QKV) / 256), 256, 0, stream>>>(
        wqkv + (size_t)l * D * QKV, wqkvT + (size_t)l * QKV * D, D, QKV);
    tconv_kernel<<<dim3((D * D) / 256), 256, 0, stream>>>(
        wout + (size_t)l * D * D, woutT + (size_t)l * D * D, D, D);
    tconv_kernel<<<dim3((D * MLP) / 256), 256, 0, stream>>>(
        w1 + (size_t)l * D * MLP, w1T + (size_t)l * MLP * D, D, MLP);
    tconv_kernel<<<dim3((MLP * D) / 256), 256, 0, stream>>>(
        w2 + (size_t)l * MLP * D, w2T + (size_t)l * D * MLP, MLP, D);
  }

  for (int l = 0; l < L; ++l) {
    ln_kernel<<<dim3(BN), 256, 0, stream>>>(
        xo, ln1w + (size_t)l * D, ln1b + (size_t)l * D, hbuf);
    gemm_bf16_kernel<0><<<dim3(QKV / 128, BN / 128), 256, 0, stream>>>(
        hbuf, wqkvT + (size_t)l * QKV * D, nullptr, nullptr, qkvb, BN, QKV, D);
    attn_kernel<<<dim3(N / 64, B * H), 128, 0, stream>>>(qkvb, attb);
    gemm_bf16_kernel<2><<<dim3(D / 128, BN / 128), 256, 0, stream>>>(
        attb, woutT + (size_t)l * D * D, bout + (size_t)l * D, xo, nullptr,
        BN, D, D);
    ln_kernel<<<dim3(BN), 256, 0, stream>>>(
        xo, ln2w + (size_t)l * D, ln2b + (size_t)l * D, hbuf);
    gemm_bf16_kernel<1><<<dim3(MLP / 128, BN / 128), 256, 0, stream>>>(
        hbuf, w1T + (size_t)l * MLP * D, b1 + (size_t)l * MLP, nullptr, mlpb,
        BN, MLP, D);
    gemm_bf16_kernel<2><<<dim3(D / 128, BN / 128), 256, 0, stream>>>(
        mlpb, w2T + (size_t)l * D * MLP, b2 + (size_t)l * D, xo, nullptr,
        BN, D, MLP);
  }
}
